// IDIR_61761629716686
// MI455X (gfx1250) — matrix operand order repaired and checked by reading, not compiled, not run on hardware
//
#include <hip/hip_runtime.h>
#include <hip/hip_bf16.h>
#include <math.h>

// ---------------- problem constants (from reference) ----------------
#define HID     512
#define NTOK    2048            // B*S = 2*1024
#define NEXP    4
#define FEXP    2048
#define RHID    2560
#define VOCABSZ 50000
#define KTOPV   32
#define MAXIT   8
#define TOLV    1e-4f
#define NH      (NTOK*HID)      // 1,048,576

typedef __attribute__((ext_vector_type(16))) __bf16 v16bf;
typedef __attribute__((ext_vector_type(8)))  __bf16 v8bf;
typedef __attribute__((ext_vector_type(4)))  __bf16 v4bf;
typedef __attribute__((ext_vector_type(8)))  float  v8f;
typedef __attribute__((ext_vector_type(4)))  float  v4f;

// =====================================================================
// bf16-WMMA GEMM: C[M,N] = A[M,K] @ B[K,N] (+bias) (+add0) (relu?)
//   A: row-major f32 (converted to bf16 while staging to LDS).
//   Bt: pre-converted bf16 weights in TRANSPOSED layout [N][K]; the LDS
//       B tile is a pure byte copy staged with global_load_async_to_lds_b128.
// Interior blocks: double-buffered software pipeline — stage tile i+1
// (async B + A cvt) while WMMA-ing tile i; ONE barrier per K-tile.
// Tile 64x64x32, 256 threads = 8 waves, each wave -> 16x32 of C (2 WMMA).
// Requires: M % 64 == 0 (M is always NTOK), K % 32 == 0 (always true).
// =====================================================================
#define TM 64
#define TN 64
#define TK 32
#define LDT 40   // LDS row stride in bf16 (80 B: 16B aligned, bank-staggered)

__device__ __forceinline__ void frag_wmma(const __bf16 (*la)[LDT],
                                          const __bf16 (*lb)[LDT],
                                          int wm, int wn, int l15, int lh,
                                          v8f& acc0, v8f& acc1)
{
    // A: lane(m=l15+wm, h=lh): k runs [8h..8h+7] and [16+8h..23+8h]
    const __bf16* Ar = &la[wm + l15][0];
    v8bf a0 = *(const v8bf*)(Ar + 8 * lh);
    v8bf a1 = *(const v8bf*)(Ar + 16 + 8 * lh);
    v16bf af = __builtin_shufflevector(a0, a1,
                0,1,2,3,4,5,6,7,8,9,10,11,12,13,14,15);
    // B: lane(n, h): k runs [8h..8h+7] and [16+8h..23+8h] (same as A)
    const __bf16* Br0 = &lb[wn + l15][8 * lh];
    v8bf b0a = *(const v8bf*)(Br0);
    v8bf b0b = *(const v8bf*)(Br0 + 16);
    v16bf bfr0 = __builtin_shufflevector(b0a, b0b,
                0,1,2,3,4,5,6,7,8,9,10,11,12,13,14,15);
    const __bf16* Br1 = &lb[wn + 16 + l15][8 * lh];
    v8bf b1a = *(const v8bf*)(Br1);
    v8bf b1b = *(const v8bf*)(Br1 + 16);
    v16bf bfr1 = __builtin_shufflevector(b1a, b1b,
                0,1,2,3,4,5,6,7,8,9,10,11,12,13,14,15);

    acc0 = __builtin_amdgcn_wmma_f32_16x16x32_bf16(false, af, false, bfr0,
                                                   (short)0, acc0, false, false);
    acc1 = __builtin_amdgcn_wmma_f32_16x16x32_bf16(false, af, false, bfr1,
                                                   (short)0, acc1, false, false);
}

__global__ __launch_bounds__(256)
void gemm_wmma(const int* gate,
               const float* __restrict__ A, const __bf16* __restrict__ Bt,
               const float* __restrict__ bias, const float* __restrict__ add0,
               float* __restrict__ C,
               int M, int N, int K, int relu)
{
    if (gate && *gate) return;
    __shared__ __align__(16) __bf16 lA[2][TM][LDT];   // lA[buf][m][k]
    __shared__ __align__(16) __bf16 lB[2][TN][LDT];   // lB[buf][n][k]

    const int bm   = blockIdx.y * TM;
    const int bn   = blockIdx.x * TN;
    const int tid  = threadIdx.x;
    const int lane = tid & 31;
    const int wave = tid >> 5;
    const int wm   = (wave & 3) << 4;   // 0,16,32,48
    const int wn   = (wave >> 2) << 5;  // 0,32
    const int l15  = lane & 15;
    const int lh   = lane >> 4;

    const bool fullN = (bn + TN <= N);

    // A-stage addressing: each thread owns 2 float4 (row r, 4 cols c4)
    const int ar0 = tid >> 3;            // 0..31
    const int ac  = (tid & 7) * 4;       // 0..28
    // B-stage addressing: each thread owns one 8xbf16 (16 B) chunk
    const int br  = tid >> 2;            // 0..63 (n within tile)
    const int bc  = (tid & 3) * 8;       // 0,8,16,24 (k within tile)

    // LDS byte offsets for async B destinations (flat LDS aperture:
    // low 32 bits of the generic pointer == LDS address)
    const unsigned ldsB0 = (unsigned)(size_t)(void*)&lB[0][br][bc];
    const unsigned ldsB1 = (unsigned)(size_t)(void*)&lB[1][br][bc];

    v8f acc0 = {}; v8f acc1 = {};

    if (fullN) {
        const int nt = K / TK;
        // stage(j, buf): async B copy + A f32->bf16 conversion into LDS
        auto stage = [&](int j, int buf) {
            int k0 = j * TK;
            const __bf16* gsrc = Bt + (size_t)(bn + br) * K + k0 + bc;
            asm volatile("global_load_async_to_lds_b128 %0, %1, off"
                         :: "v"(buf ? ldsB1 : ldsB0),
                            "v"((unsigned long long)(size_t)gsrc)
                         : "memory");
            if (k0 + TK < K)   // prefetch A one further tile ahead
                __builtin_prefetch(A + (size_t)(bm + ar0) * K + (k0 + TK) + ac, 0, 1);
            #pragma unroll
            for (int i = 0; i < 2; ++i) {
                int r = ar0 + i * 32;
                v4f v = *(const v4f*)(A + (size_t)(bm + r) * K + k0 + ac);
                v4bf w;
                w[0] = (__bf16)v[0]; w[1] = (__bf16)v[1];
                w[2] = (__bf16)v[2]; w[3] = (__bf16)v[3];
                *(v4bf*)&lA[buf][r][ac] = w;
            }
        };

        stage(0, 0);
        asm volatile("s_wait_asynccnt 0x0" ::: "memory");
        __syncthreads();

        for (int i = 0; i < nt; ++i) {
            int cur = i & 1;
            if (i + 1 < nt) stage(i + 1, (i + 1) & 1);  // overlap with WMMA below
            frag_wmma(lA[cur], lB[cur], wm, wn, l15, lh, acc0, acc1);
            asm volatile("s_wait_asynccnt 0x0" ::: "memory");
            __syncthreads();     // one barrier per K-tile
        }
    } else {
        // ragged-N edge (last column tile of the logits GEMM): simple path
        for (int k0 = 0; k0 < K; k0 += TK) {
            v8bf v = {};
            if (bn + br < N)
                v = *(const v8bf*)(Bt + (size_t)(bn + br) * K + k0 + bc);
            *(v8bf*)&lB[0][br][bc] = v;
            #pragma unroll
            for (int i = 0; i < 2; ++i) {
                int r = ar0 + i * 32;
                v4f av = *(const v4f*)(A + (size_t)(bm + r) * K + k0 + ac);
                v4bf w;
                w[0] = (__bf16)av[0]; w[1] = (__bf16)av[1];
                w[2] = (__bf16)av[2]; w[3] = (__bf16)av[3];
                *(v4bf*)&lA[0][r][ac] = w;
            }
            __syncthreads();
            frag_wmma(lA[0], lB[0], wm, wn, l15, lh, acc0, acc1);
            __syncthreads();
        }
    }

    // ---- epilogue: C layout lane=col(0..15), vgpr v = row v+8*lh ----
    if (fullN) {
        #pragma unroll
        for (int v = 0; v < 8; ++v) {
            int row = bm + wm + v + 8 * lh;
            int c0  = bn + wn + l15;
            int c1  = c0 + 16;
            float x0 = acc0[v], x1 = acc1[v];
            if (bias) { x0 += bias[c0]; x1 += bias[c1]; }
            if (add0) { x0 += add0[(size_t)row * N + c0];
                        x1 += add0[(size_t)row * N + c1]; }
            if (relu) { x0 = fmaxf(x0, 0.f); x1 = fmaxf(x1, 0.f); }
            C[(size_t)row * N + c0] = x0;
            C[(size_t)row * N + c1] = x1;
        }
    } else {
        #pragma unroll
        for (int v = 0; v < 8; ++v) {
            int row = bm + wm + v + 8 * lh;
            int c0  = bn + wn + l15;
            int c1  = c0 + 16;
            if (c0 < N) {
                float x = acc0[v];
                if (bias) x += bias[c0];
                if (add0) x += add0[(size_t)row * N + c0];
                if (relu) x = fmaxf(x, 0.f);
                C[(size_t)row * N + c0] = x;
            }
            if (c1 < N) {
                float x = acc1[v];
                if (bias) x += bias[c1];
                if (add0) x += add0[(size_t)row * N + c1];
                if (relu) x = fmaxf(x, 0.f);
                C[(size_t)row * N + c1] = x;
            }
        }
    }
}

// =====================================================================
// Weight preparation: f32 -> bf16, plain or transposed into [N][K].
// =====================================================================
__global__ void convert_bf16_kernel(const float* __restrict__ src,
                                    __bf16* __restrict__ dst, size_t n)
{
    size_t i = (size_t)blockIdx.x * 256 + threadIdx.x;
    if (i < n) dst[i] = (__bf16)src[i];
}

// dst[n*K + k] = src[k*N + n]
__global__ void transpose_bf16_kernel(const float* __restrict__ src,
                                      __bf16* __restrict__ dst, int K, int N)
{
    size_t i = (size_t)blockIdx.x * 256 + threadIdx.x;
    size_t total = (size_t)K * N;
    if (i >= total) return;
    size_t n = i / K, k = i - n * K;
    dst[i] = (__bf16)src[k * (size_t)N + n];
}

// =====================================================================
// LayerNorm over last dim D (one block per row), optional relu-in/out.
// =====================================================================
__global__ __launch_bounds__(256)
void ln_kernel(const int* gate, const float* __restrict__ in, float* __restrict__ out,
               const float* __restrict__ w, const float* __restrict__ b,
               int D, int relu_in, int relu_out)
{
    if (gate && *gate) return;
    const int row = blockIdx.x;
    const float* x = in + (size_t)row * D;
    float* y = out + (size_t)row * D;
    __shared__ float red[256];
    const int t = threadIdx.x;

    float s = 0.f, s2 = 0.f;
    for (int d = t; d < D; d += 256) {
        float v = x[d]; if (relu_in) v = fmaxf(v, 0.f);
        s += v; s2 += v * v;
    }
    red[t] = s; __syncthreads();
    for (int o = 128; o; o >>= 1) { if (t < o) red[t] += red[t + o]; __syncthreads(); }
    float mean = red[0] / (float)D;
    __syncthreads();
    red[t] = s2; __syncthreads();
    for (int o = 128; o; o >>= 1) { if (t < o) red[t] += red[t + o]; __syncthreads(); }
    float var = red[0] / (float)D - mean * mean;
    float rstd = rsqrtf(var + 1e-5f);

    for (int d = t; d < D; d += 256) {
        float v = x[d]; if (relu_in) v = fmaxf(v, 0.f);
        float o2 = (v - mean) * rstd * w[d] + b[d];
        if (relu_out) o2 = fmaxf(o2, 0.f);
        y[d] = o2;
    }
}

// =====================================================================
// Router: hn = LN(h); g = softmax(hn @ router_w); top2 -> tw (renorm), ti
// =====================================================================
__global__ __launch_bounds__(64)
void router_kernel(const int* gate, const float* __restrict__ h,
                   const float* __restrict__ lw, const float* __restrict__ lb,
                   const float* __restrict__ rw,
                   float* __restrict__ tw, int* __restrict__ ti)
{
    if (gate && *gate) return;
    const int n = blockIdx.x;
    const float* x = h + (size_t)n * HID;
    __shared__ float red[64];
    __shared__ float sg[NEXP];
    const int t = threadIdx.x;

    float s = 0.f, s2 = 0.f;
    for (int d = t; d < HID; d += 64) { float v = x[d]; s += v; s2 += v * v; }
    red[t] = s; __syncthreads();
    for (int o = 32; o; o >>= 1) { if (t < o) red[t] += red[t + o]; __syncthreads(); }
    float mean = red[0] / (float)HID; __syncthreads();
    red[t] = s2; __syncthreads();
    for (int o = 32; o; o >>= 1) { if (t < o) red[t] += red[t + o]; __syncthreads(); }
    float rstd = rsqrtf(red[0] / (float)HID - mean * mean + 1e-5f);
    __syncthreads();

    float acc[NEXP] = {0.f, 0.f, 0.f, 0.f};
    for (int d = t; d < HID; d += 64) {
        float hn = (x[d] - mean) * rstd * lw[d] + lb[d];
        #pragma unroll
        for (int e = 0; e < NEXP; ++e) acc[e] += hn * rw[d * NEXP + e];
    }
    for (int e = 0; e < NEXP; ++e) {
        red[t] = acc[e]; __syncthreads();
        for (int o = 32; o; o >>= 1) { if (t < o) red[t] += red[t + o]; __syncthreads(); }
        if (t == 0) sg[e] = red[0];
        __syncthreads();
    }
    if (t == 0) {
        float mx = sg[0];
        for (int e = 1; e < NEXP; ++e) mx = fmaxf(mx, sg[e]);
        float g[NEXP], sum = 0.f;
        for (int e = 0; e < NEXP; ++e) { g[e] = expf(sg[e] - mx); sum += g[e]; }
        for (int e = 0; e < NEXP; ++e) g[e] /= sum;
        int i0 = 0;
        for (int e = 1; e < NEXP; ++e) if (g[e] > g[i0]) i0 = e;
        int i1 = -1;
        for (int e = 0; e < NEXP; ++e) {
            if (e == i0) continue;
            if (i1 < 0 || g[e] > g[i1]) i1 = e;
        }
        float w0 = g[i0], w1 = g[i1];
        float den = w0 + w1 + 1e-8f;
        tw[2 * n] = w0 / den; tw[2 * n + 1] = w1 / den;
        ti[2 * n] = i0;       ti[2 * n + 1] = i1;
    }
}

// =====================================================================
// Top-32 soft mask over 512 scores (tie -> lower index), softmax over
// the 32 kept values, z *= scattered weights.
// =====================================================================
__global__ __launch_bounds__(256)
void topk_mask_kernel(const int* gate, const float* __restrict__ scores,
                      float* __restrict__ z)
{
    if (gate && *gate) return;
    const int n = blockIdx.x;
    __shared__ float s[HID];
    __shared__ float wsel[HID];
    __shared__ float rv[256];
    __shared__ int   ri[256];
    __shared__ float topv[KTOPV];
    __shared__ int   topi[KTOPV];
    const int t = threadIdx.x;

    s[t]       = scores[(size_t)n * HID + t];
    s[t + 256] = scores[(size_t)n * HID + t + 256];
    wsel[t] = 0.f; wsel[t + 256] = 0.f;
    __syncthreads();

    for (int it = 0; it < KTOPV; ++it) {
        float v0 = s[t], v1 = s[t + 256];
        float bv; int bi;
        if (v1 > v0) { bv = v1; bi = t + 256; } else { bv = v0; bi = t; }
        rv[t] = bv; ri[t] = bi;
        __syncthreads();
        for (int o = 128; o; o >>= 1) {
            if (t < o) {
                float ov = rv[t + o]; int oi = ri[t + o];
                if (ov > rv[t] || (ov == rv[t] && oi < ri[t])) { rv[t] = ov; ri[t] = oi; }
            }
            __syncthreads();
        }
        if (t == 0) { topv[it] = rv[0]; topi[it] = ri[0]; s[ri[0]] = -3.0e38f; }
        __syncthreads();
    }
    if (t == 0) {
        float mx = topv[0];
        float sum = 0.f;
        for (int i = 0; i < KTOPV; ++i) { float e = expf(topv[i] - mx); topv[i] = e; sum += e; }
        for (int i = 0; i < KTOPV; ++i) wsel[topi[i]] = topv[i] / sum;
    }
    __syncthreads();
    z[(size_t)n * HID + t]       *= wsel[t];
    z[(size_t)n * HID + t + 256] *= wsel[t + 256];
}

// =====================================================================
// Spectral normalization of Wh (512x512): 12 power iterations, one WGP.
// =====================================================================
__global__ __launch_bounds__(512)
void spectral_norm_kernel(const float* __restrict__ W, float* __restrict__ Wsn)
{
    __shared__ float u[HID], v[HID], red[HID];
    const int t = threadIdx.x;
    u[t] = 0.044194173824159216f;  // 1/sqrt(512)
    __syncthreads();

    for (int it = 0; it < 12; ++it) {
        float s = 0.f;
        for (int k = 0; k < HID; ++k) s += W[(size_t)t * HID + k] * u[k];
        red[t] = s * s; __syncthreads();
        for (int o = 256; o; o >>= 1) { if (t < o) red[t] += red[t + o]; __syncthreads(); }
        float nv = sqrtf(red[0]) + 1e-12f;
        __syncthreads();
        v[t] = s / nv; __syncthreads();

        float s2 = 0.f;
        for (int i = 0; i < HID; ++i) s2 += v[i] * W[(size_t)i * HID + t];
        red[t] = s2 * s2; __syncthreads();
        for (int o = 256; o; o >>= 1) { if (t < o) red[t] += red[t + o]; __syncthreads(); }
        float nu = sqrtf(red[0]) + 1e-12f;
        __syncthreads();
        u[t] = s2 / nu; __syncthreads();
    }
    float s = 0.f;
    for (int k = 0; k < HID; ++k) s += W[(size_t)t * HID + k] * u[k];
    red[t] = v[t] * s; __syncthreads();
    for (int o = 256; o; o >>= 1) { if (t < o) red[t] += red[t + o]; __syncthreads(); }
    float inv = 1.0f / red[0];
    __syncthreads();
    for (size_t i = t; i < (size_t)HID * HID; i += HID) Wsn[i] = W[i] * inv;
}

// ---------------- small elementwise kernels ----------------
__global__ void embed_gather_kernel(const int* __restrict__ x, const float* __restrict__ emb,
                                    float* __restrict__ e)
{
    int n = blockIdx.x;
    int tok = x[n];
    for (int d = threadIdx.x; d < HID; d += 256)
        e[(size_t)n * HID + d] = emb[(size_t)tok * HID + d];
}

__global__ void copy_kernel(const int* gate, const float* __restrict__ a,
                            float* __restrict__ b, int n)
{
    if (gate && *gate) return;
    int i = blockIdx.x * 256 + threadIdx.x;
    if (i < n) b[i] = a[i];
}

__global__ void concat_kernel(const int* gate, const float* __restrict__ z,
                              const float* __restrict__ h, float* __restrict__ cat)
{
    if (gate && *gate) return;
    size_t i = (size_t)blockIdx.x * 256 + threadIdx.x;   // over N*HID
    size_t n = i >> 9, d = i & (HID - 1);
    cat[n * (2 * HID) + d]       = z[i];
    cat[n * (2 * HID) + HID + d] = h[i];
}

__global__ void add3_kernel(const int* gate, const float* __restrict__ a,
                            const float* __restrict__ b, const float* __restrict__ c,
                            float* __restrict__ o)
{
    if (gate && *gate) return;
    size_t i = (size_t)blockIdx.x * 256 + threadIdx.x;
    o[i] = a[i] + b[i] + c[i];
}

__global__ void expert_accum_kernel(const int* gate, const float* __restrict__ eo,
                                    const float* __restrict__ tw, const int* __restrict__ ti,
                                    int e, float* __restrict__ out)
{
    if (gate && *gate) return;
    size_t i = (size_t)blockIdx.x * 256 + threadIdx.x;
    int n = (int)(i >> 9);
    float w = 0.f;
    if (ti[2 * n] == e)     w += tw[2 * n];
    if (ti[2 * n + 1] == e) w += tw[2 * n + 1];
    out[i] += w * eo[i];
}

__global__ void diff_reduce_kernel(const int* gate, const float* __restrict__ a,
                                   const float* __restrict__ b, float* accum, int n)
{
    if (gate && *gate) return;
    __shared__ float red[256];
    int i = blockIdx.x * 256 + threadIdx.x;
    float d = 0.f;
    if (i < n) { float t2 = a[i] - b[i]; d = t2 * t2; }
    red[threadIdx.x] = d; __syncthreads();
    for (int o = 128; o; o >>= 1) { if (threadIdx.x < o) red[threadIdx.x] += red[threadIdx.x + o]; __syncthreads(); }
    if (threadIdx.x == 0) atomicAdd(accum, red[0]);
}

__global__ void flags_update_kernel(int* flags, const float* accum, float tol)
{
    if (flags[0]) return;
    flags[1] += 1;
    if (sqrtf(*accum) < tol) flags[0] = 1;
}

__global__ void write_iters_kernel(float* out, const int* flags)
{
    out[(size_t)NTOK * VOCABSZ] = (float)flags[1];
}

// =====================================================================
// Host-side orchestration
// =====================================================================
struct Ptrs {
    // parameters (f32 inputs)
    const float *emb, *out_b, *Wh, *Wx, *core_ln_w, *core_ln_b;
    const float *router_w, *router_ln_w, *router_ln_b;
    const float *exp_W1, *exp_ln_w, *exp_ln_b, *exp_W2, *exp_out_ln_w, *exp_out_ln_b;
    const float *T1_w, *T1_b, *T_ln_w, *T_ln_b, *T2_w, *T2_b;
    const float *R1_w, *R1_b, *R_ln_w, *R_ln_b, *R2_w, *R2_b, *rsn_ln_w, *rsn_ln_b;
    // f32 workspace
    float *wh_sn, *e, *exWx, *H, *Hn, *pre, *z, *eo, *eout, *efin, *zr, *cat, *t, *scores, *rsn;
    float *tw, *accum;
    int   *ti, *flags;
    // bf16 transposed weights [N][K]
    __bf16 *whT, *wxT, *e1T, *e2T, *t1T, *t2T, *r1T, *r2T, *embB;
};

static inline void launch_gemm(hipStream_t s, const int* gate, const float* A,
                               const __bf16* Bt, const float* bias, const float* add0,
                               float* C, int M, int N, int K, int relu)
{
    dim3 g((N + TN - 1) / TN, (M + TM - 1) / TM);
    gemm_wmma<<<g, 256, 0, s>>>(gate, A, Bt, bias, add0, C, M, N, K, relu);
}

static inline void launch_transpose(hipStream_t s, const float* src, __bf16* dst, int K, int N)
{
    size_t total = (size_t)K * N;
    transpose_bf16_kernel<<<(unsigned)((total + 255) / 256), 256, 0, s>>>(src, dst, K, N);
}

static void launch_step(hipStream_t s, const int* gate, const Ptrs& P, float* hout)
{
    const int EL = NH / 256;
    // ---- MoE ----
    router_kernel<<<NTOK, 64, 0, s>>>(gate, P.H, P.router_ln_w, P.router_ln_b, P.router_w, P.tw, P.ti);
    hipMemsetAsync(P.eout, 0, (size_t)NH * sizeof(float), s);
    for (int ex = 0; ex < NEXP; ++ex) {
        launch_gemm(s, gate, P.H, P.e1T + (size_t)ex * HID * FEXP, nullptr, nullptr,
                    P.z, NTOK, FEXP, HID, 0);
        ln_kernel<<<NTOK, 256, 0, s>>>(gate, P.z, P.z, P.exp_ln_w + (size_t)ex * FEXP,
                                       P.exp_ln_b + (size_t)ex * FEXP, FEXP, 0, 1);
        launch_gemm(s, gate, P.z, P.e2T + (size_t)ex * FEXP * HID, nullptr, nullptr,
                    P.eo, NTOK, HID, FEXP, 0);
        expert_accum_kernel<<<EL, 256, 0, s>>>(gate, P.eo, P.tw, P.ti, ex, P.eout);
    }
    ln_kernel<<<NTOK, 256, 0, s>>>(gate, P.eout, P.efin, P.exp_out_ln_w, P.exp_out_ln_b, HID, 0, 0);

    // ---- reasoning ----
    copy_kernel<<<EL, 256, 0, s>>>(gate, P.H, P.zr, NH);
    for (int st = 0; st < 3; ++st) {
        concat_kernel<<<EL, 256, 0, s>>>(gate, P.zr, P.H, P.cat);
        launch_gemm(s, gate, P.cat, P.t1T, P.T1_b, nullptr, P.t, NTOK, RHID, 2 * HID, 0);
        ln_kernel<<<NTOK, 256, 0, s>>>(gate, P.t, P.t, P.T_ln_w, P.T_ln_b, RHID, 0, 1);
        launch_gemm(s, gate, P.t, P.t2T, P.T2_b, nullptr, P.scores, NTOK, HID, RHID, 0);
        topk_mask_kernel<<<NTOK, 256, 0, s>>>(gate, P.scores, P.zr);
    }
    launch_gemm(s, gate, P.zr, P.r1T, P.R1_b, nullptr, P.t, NTOK, RHID, HID, 0);
    ln_kernel<<<NTOK, 256, 0, s>>>(gate, P.t, P.t, P.R_ln_w, P.R_ln_b, RHID, 0, 1);
    launch_gemm(s, gate, P.t, P.r2T, P.R2_b, nullptr, P.rsn, NTOK, HID, RHID, 0);
    ln_kernel<<<NTOK, 256, 0, s>>>(gate, P.rsn, P.rsn, P.rsn_ln_w, P.rsn_ln_b, HID, 0, 0);

    // ---- combine: LN(relu(h@Wh_sn + e@Wx + expert + reasoning)) ----
    add3_kernel<<<EL, 256, 0, s>>>(gate, P.exWx, P.efin, P.rsn, P.pre);
    launch_gemm(s, gate, P.H, P.whT, nullptr, P.pre, hout, NTOK, HID, HID, 1);
    ln_kernel<<<NTOK, 256, 0, s>>>(gate, hout, hout, P.core_ln_w, P.core_ln_b, HID, 0, 0);
}

extern "C" void kernel_launch(void* const* d_in, const int* in_sizes, int n_in,
                              void* d_out, int out_size, void* d_ws, size_t ws_size,
                              hipStream_t stream)
{
    (void)in_sizes; (void)n_in; (void)ws_size;
    const int* x = (const int*)d_in[0];
    Ptrs P;
    P.emb          = (const float*)d_in[1];
    P.out_b        = (const float*)d_in[2];
    P.Wh           = (const float*)d_in[3];
    P.Wx           = (const float*)d_in[4];
    P.core_ln_w    = (const float*)d_in[5];
    P.core_ln_b    = (const float*)d_in[6];
    P.router_w     = (const float*)d_in[7];
    P.router_ln_w  = (const float*)d_in[8];
    P.router_ln_b  = (const float*)d_in[9];
    P.exp_W1       = (const float*)d_in[10];
    P.exp_ln_w     = (const float*)d_in[11];
    P.exp_ln_b     = (const float*)d_in[12];
    P.exp_W2       = (const float*)d_in[13];
    P.exp_out_ln_w = (const float*)d_in[14];
    P.exp_out_ln_b = (const float*)d_in[15];
    P.T1_w = (const float*)d_in[16]; P.T1_b = (const float*)d_in[17];
    P.T_ln_w = (const float*)d_in[18]; P.T_ln_b = (const float*)d_in[19];
    P.T2_w = (const float*)d_in[20]; P.T2_b = (const float*)d_in[21];
    P.R1_w = (const float*)d_in[22]; P.R1_b = (const float*)d_in[23];
    P.R_ln_w = (const float*)d_in[24]; P.R_ln_b = (const float*)d_in[25];
    P.R2_w = (const float*)d_in[26]; P.R2_b = (const float*)d_in[27];
    P.rsn_ln_w = (const float*)d_in[28]; P.rsn_ln_b = (const float*)d_in[29];

    // ---- carve workspace (256B aligned allocations) ----
    char* base = (char*)d_ws;
    size_t off = 0;
    auto alloc = [&](size_t nbytes) -> void* {
        void* p = base + off;
        off += (nbytes + 255) & ~(size_t)255;
        return p;
    };
    P.wh_sn  = (float*)alloc((size_t)HID * HID * 4);
    P.e      = (float*)alloc((size_t)NH * 4);
    P.exWx   = (float*)alloc((size_t)NH * 4);
    P.H      = (float*)alloc((size_t)NH * 4);
    P.Hn     = (float*)alloc((size_t)NH * 4);
    P.pre    = (float*)alloc((size_t)NH * 4);
    P.eo     = (float*)alloc((size_t)NH * 4);
    P.eout   = (float*)alloc((size_t)NH * 4);
    P.efin   = (float*)alloc((size_t)NH * 4);
    P.zr     = (float*)alloc((size_t)NH * 4);
    P.scores = (float*)alloc((size_t)NH * 4);
    P.rsn    = (float*)alloc((size_t)NH * 4);
    P.z      = (float*)alloc((size_t)NTOK * FEXP * 4);
    P.cat    = (float*)alloc((size_t)NTOK * 2 * HID * 4);
    P.t      = (float*)alloc((size_t)NTOK * RHID * 4);
    P.tw     = (float*)alloc((size_t)NTOK * 2 * 4);
    P.ti     = (int*)  alloc((size_t)NTOK * 2 * 4);
    P.flags  = (int*)  alloc(256);
    P.accum  = (float*)alloc(256);
    // bf16 transposed weights [N][K]
    P.whT  = (__bf16*)alloc((size_t)HID * HID * 2);
    P.wxT  = (__bf16*)alloc((size_t)HID * HID * 2);
    P.e1T  = (__bf16*)alloc((size_t)NEXP * HID * FEXP * 2);
    P.e2T  = (__bf16*)alloc((size_t)NEXP * FEXP * HID * 2);
    P.t1T  = (__bf16*)alloc((size_t)2 * HID * RHID * 2);
    P.t2T  = (__bf16*)alloc((size_t)RHID * HID * 2);
    P.r1T  = (__bf16*)alloc((size_t)HID * RHID * 2);
    P.r2T  = (__bf16*)alloc((size_t)RHID * HID * 2);
    P.embB = (__bf16*)alloc((size_t)VOCABSZ * HID * 2);

    float* out = (float*)d_out;

    // ---- one-time weight prep (iteration-invariant) ----
    spectral_norm_kernel<<<1, 512, 0, stream>>>(P.Wh, P.wh_sn);
    launch_transpose(stream, P.wh_sn, P.whT, HID, HID);
    launch_transpose(stream, P.Wx,    P.wxT, HID, HID);
    for (int ex = 0; ex < NEXP; ++ex) {
        launch_transpose(stream, P.exp_W1 + (size_t)ex * HID * FEXP,
                         P.e1T + (size_t)ex * HID * FEXP, HID, FEXP);
        launch_transpose(stream, P.exp_W2 + (size_t)ex * FEXP * HID,
                         P.e2T + (size_t)ex * FEXP * HID, FEXP, HID);
    }
    launch_transpose(stream, P.T1_w, P.t1T, 2 * HID, RHID);
    launch_transpose(stream, P.T2_w, P.t2T, RHID, HID);
    launch_transpose(stream, P.R1_w, P.r1T, HID, RHID);
    launch_transpose(stream, P.R2_w, P.r2T, RHID, HID);
    {   // emb: h @ emb^T wants B[k][n]=emb[n][k] -> [N][K] layout == emb itself
        size_t total = (size_t)VOCABSZ * HID;
        convert_bf16_kernel<<<(unsigned)((total + 255) / 256), 256, 0, stream>>>(P.emb, P.embB, total);
    }

    embed_gather_kernel<<<NTOK, 256, 0, stream>>>(x, P.emb, P.e);
    launch_gemm(stream, nullptr, P.e, P.wxT, nullptr, nullptr, P.exWx, NTOK, HID, HID, 0);
    hipMemsetAsync(P.H, 0, (size_t)NH * sizeof(float), stream);
    hipMemsetAsync(P.flags, 0, 2 * sizeof(int), stream);

    // ---- fixed-point loop: 8 gated iterations (device-side convergence) ----
    const int EL = NH / 256;
    for (int it = 0; it < MAXIT; ++it) {
        launch_step(stream, P.flags, P, P.Hn);
        hipMemsetAsync(P.accum, 0, sizeof(float), stream);
        diff_reduce_kernel<<<EL, 256, 0, stream>>>(P.flags, P.Hn, P.H, P.accum, NH);
        copy_kernel<<<EL, 256, 0, stream>>>(P.flags, P.Hn, P.H, NH);
        flags_update_kernel<<<1, 1, 0, stream>>>(P.flags, P.accum, TOLV);
    }

    // ---- final (ungated) step: h_star ----
    launch_step(stream, nullptr, P, P.Hn);

    // ---- logits = h_star @ emb^T + out_b ----
    launch_gemm(stream, nullptr, P.Hn, P.embB, P.out_b, nullptr, out,
                NTOK, VOCABSZ, HID, 0);

    // ---- iters as trailing float output ----
    if (out_size > NTOK * VOCABSZ)
        write_iters_kernel<<<1, 1, 0, stream>>>(out, P.flags);
}
